// EMHA_74715251081263
// MI455X (gfx1250) — matrix operand order repaired and checked by reading, not compiled, not run on hardware
//
#include <hip/hip_runtime.h>
#include <hip/hip_bf16.h>

// CDNA5 / gfx1250, wave32.
// - all GEMMs + attention through v_wmma_f32_16x16x32_bf16 (f32 accum)
// - LDS tile staging via GLOBAL_LOAD_ASYNC_TO_LDS_B128 (ASYNCcnt), double buffered
// - every WMMA fragment is a ds_load_b128

typedef __attribute__((ext_vector_type(16))) __bf16 v16bf;
typedef __attribute__((ext_vector_type(8)))  float  v8f;

union BFrag { v16bf v; uint4 q[2]; };

#define BM 128
#define BN 128
#define BK 32

// ---- gfx1250 async global->LDS copy (16B per lane), ASYNCcnt tracked ----
__device__ __forceinline__ void async_copy16(const void* gptr, void* lptr) {
  uint32_t lds_off = (uint32_t)(uintptr_t)lptr;          // flat LDS addr[31:0]
  uint64_t gaddr   = (uint64_t)(uintptr_t)gptr;
  asm volatile("global_load_async_to_lds_b128 %0, %1, off"
               :: "v"(lds_off), "v"(gaddr) : "memory");
}
__device__ __forceinline__ void async_wait0() {
  asm volatile("s_wait_asynccnt 0" ::: "memory");
}

// ---------------- fp32 -> bf16 conversion (no transpose) ----------------
__global__ void f2bf_kernel(const float* __restrict__ in, __bf16* __restrict__ out,
                            long long n) {
  long long i = (long long)blockIdx.x * blockDim.x + threadIdx.x;
  long long stride = (long long)gridDim.x * blockDim.x;
  for (; i < n; i += stride) out[i] = (__bf16)in[i];
}

// ---------------- fp32 -> bf16 with per-batch transpose ----------------
// in: [bat][R][Ccols] f32  ->  out: [bat][Ccols][R] bf16   (R, Ccols multiples of 32)
__global__ __launch_bounds__(256)
void f2bf_transpose_kernel(const float* __restrict__ in, __bf16* __restrict__ out,
                           int R, int Ccols) {
  __shared__ __bf16 tile[32][33];
  const int c0 = blockIdx.x * 32;
  const int r0 = blockIdx.y * 32;
  const int bat = blockIdx.z;
  const float* ib = in + (long long)bat * R * Ccols;
  __bf16* ob = out + (long long)bat * R * Ccols;
  const int tx = threadIdx.x & 31, ty = threadIdx.x >> 5;   // 32 x 8
  #pragma unroll
  for (int j = 0; j < 4; ++j) {
    int r = r0 + ty + j * 8;
    tile[ty + j * 8][tx] = (__bf16)ib[(long long)r * Ccols + c0 + tx];
  }
  __syncthreads();
  #pragma unroll
  for (int j = 0; j < 4; ++j) {
    int c = c0 + ty + j * 8;
    ob[(long long)c * R + r0 + tx] = tile[tx][ty + j * 8];
  }
}

// ---------------- batched bf16 WMMA GEMM, async double-buffered ----------------
// C = A[M,K] * B^T (B stored [N][K], row stride ldb)  (+bias)
// BIAS: 0 none, 1 bias[m], 2 bias[n].  CT: store C transposed as [N][M] (ldc = M-stride).
template<int BIAS, bool CT, typename OutT>
__global__ __launch_bounds__(256)
void gemm_bf16_wmma(const __bf16* __restrict__ A, const __bf16* __restrict__ B,
                    const float* __restrict__ bias, OutT* __restrict__ C,
                    int M, int N, int K, int lda, int ldb, int ldc,
                    long long bsA, long long bsB, long long bsC) {
  __shared__ __align__(16) __bf16 As[2][BM][BK + 8];   // [m][k], row 80B
  __shared__ __align__(16) __bf16 Bs[2][BN][BK + 8];   // [n][k], row 80B

  const int tid  = threadIdx.x;
  const int lane = tid & 31;
  const int wave = tid >> 5;
  const int wm0  = (wave >> 1) * 32;
  const int wn0  = (wave & 1) * 64;
  const int m0   = blockIdx.y * BM;
  const int n0   = blockIdx.x * BN;
  const int bat  = blockIdx.z;

  const __bf16* Ab = A + bsA * bat;
  const __bf16* Bb = B + bsB * bat;

  v8f acc[2][4];
  #pragma unroll
  for (int i = 0; i < 2; ++i)
    #pragma unroll
    for (int j = 0; j < 4; ++j)
      #pragma unroll
      for (int r = 0; r < 8; ++r) acc[i][j][r] = 0.0f;

  const int cm    = lane & 15;
  const int khalf = (lane >> 4) * 8;
  const int kb16  = (lane >> 4) * 16;

  // async-stage one K-tile into buffer `buf`
  auto stage = [&](int buf, int kt) {
    #pragma unroll
    for (int it = 0; it < 2; ++it) {
      int idx = tid + it * 256;          // 0..511
      int m = idx >> 2, c = (idx & 3) * 8;
      int gm = m0 + m; if (gm >= M) gm = M - 1;
      async_copy16(Ab + (long long)gm * lda + kt + c, &As[buf][m][c]);
    }
    #pragma unroll
    for (int it = 0; it < 2; ++it) {
      int idx = tid + it * 256;
      int n = idx >> 2, c = (idx & 3) * 8;
      async_copy16(Bb + (long long)(n0 + n) * ldb + kt + c, &Bs[buf][n][c]);
    }
  };

  stage(0, 0);
  async_wait0();
  __syncthreads();

  const int nk = K / BK;
  for (int ik = 0; ik < nk; ++ik) {
    const int cur = ik & 1;
    if (ik + 1 < nk) stage(cur ^ 1, (ik + 1) * BK);   // overlap next tile fetch

    BFrag af[2];
    #pragma unroll
    for (int mi = 0; mi < 2; ++mi) {
      const int row = wm0 + mi * 16 + cm;
      af[mi].q[0] = *(const uint4*)(&As[cur][row][khalf]);
      af[mi].q[1] = *(const uint4*)(&As[cur][row][khalf + 16]);
    }
    BFrag bfr[4];
    #pragma unroll
    for (int ni = 0; ni < 4; ++ni) {
      const int col = wn0 + ni * 16 + cm;
      bfr[ni].q[0] = *(const uint4*)(&Bs[cur][col][khalf]);
      bfr[ni].q[1] = *(const uint4*)(&Bs[cur][col][khalf + 16]);
    }
    #pragma unroll
    for (int mi = 0; mi < 2; ++mi)
      #pragma unroll
      for (int ni = 0; ni < 4; ++ni)
        acc[mi][ni] = __builtin_amdgcn_wmma_f32_16x16x32_bf16(
            false, af[mi].v, false, bfr[ni].v, (short)0, acc[mi][ni], false, false);

    if (ik + 1 < nk) async_wait0();
    __syncthreads();
  }

  // ---- epilogue ----
  const int hs8 = (lane >> 4) * 8;
  if constexpr (CT) {
    // store C^T: [n][m], m contiguous -> packed b128 stores
    #pragma unroll
    for (int mi = 0; mi < 2; ++mi) {
      const int mb = m0 + wm0 + mi * 16 + hs8;
      if (mb < M) {
        #pragma unroll
        for (int ni = 0; ni < 4; ++ni) {
          const int n = n0 + wn0 + ni * 16 + cm;
          union { __bf16 h[8]; uint4 q; } pk;
          #pragma unroll
          for (int r = 0; r < 8; ++r) {
            float v = acc[mi][ni][r];
            if (BIAS == 1) v += bias[mb + r];
            if (BIAS == 2) v += bias[n];
            pk.h[r] = (__bf16)v;
          }
          *(uint4*)(&C[bsC * bat + (long long)n * ldc + mb]) = pk.q;
        }
      }
    }
  } else {
    #pragma unroll
    for (int mi = 0; mi < 2; ++mi) {
      #pragma unroll
      for (int ni = 0; ni < 4; ++ni) {
        #pragma unroll
        for (int r = 0; r < 8; ++r) {
          int m = m0 + wm0 + mi * 16 + hs8 + r;
          int n = n0 + wn0 + ni * 16 + cm;
          if (m < M) {
            float v = acc[mi][ni][r];
            if (BIAS == 1) v += bias[m];
            if (BIAS == 2) v += bias[n];
            C[bsC * bat + (long long)m * ldc + n] = (OutT)v;
          }
        }
      }
    }
  }
}

// ---------------- fused attention: one (b,h,s) block per CTA ----------------
// qkv: [B][288][1536] bf16 (q cols 0..511, k 512..1023, v 1024..1535)
// out: [B][288][512] bf16, out[b][h*36+dd][s*128+n]
__global__ __launch_bounds__(256)
void attn_wmma_kernel(const __bf16* __restrict__ qkv, __bf16* __restrict__ outp) {
  // Qs[128][72] (n-major, dd-contig) | Ks[128][72] | Vt[64][136] (dd-major, m-contig)
  // Ps[128][136] aliases Qs+Ks after S-phase.
  __shared__ __align__(16) char smem[18432 * 2 + 17408];   // 54272 B
  __bf16 (*Qs)[72]  = (__bf16(*)[72])(smem);
  __bf16 (*Ks)[72]  = (__bf16(*)[72])(smem + 18432);
  __bf16 (*Vt)[136] = (__bf16(*)[136])(smem + 36864);
  __bf16 (*Ps)[136] = (__bf16(*)[136])(smem);

  const int tid = threadIdx.x, lane = tid & 31, wave = tid >> 5;
  const int s = blockIdx.x, h = blockIdx.y, b = blockIdx.z;
  const long long rowbase = ((long long)b * 288 + h * 36) * 1536;
  const int colq = s * 128;

  // zero-fill all of LDS (covers dd 36..63 padding)
  {
    uint4 z = {0, 0, 0, 0};
    for (int idx = tid; idx < (18432 * 2 + 17408) / 16; idx += 256)
      *(uint4*)(smem + idx * 16) = z;
  }
  __syncthreads();

  // V: async straight copy into [dd][m]; Q/K: transpose-scatter into [n][dd]
  for (int idx = tid; idx < 36 * 16; idx += 256) {
    int dd = idx >> 4, ch = (idx & 15) * 8;
    const __bf16* p = qkv + rowbase + (long long)dd * 1536 + colq + ch;
    async_copy16(p + 1024, &Vt[dd][ch]);
    uint4 qv = *(const uint4*)(p);
    uint4 kv = *(const uint4*)(p + 512);
    const __bf16* pq = (const __bf16*)&qv;
    const __bf16* pk = (const __bf16*)&kv;
    #pragma unroll
    for (int e = 0; e < 8; ++e) {
      Qs[ch + e][dd] = pq[e];
      Ks[ch + e][dd] = pk[e];
    }
  }
  async_wait0();
  __syncthreads();

  const int cm = lane & 15, hs = lane >> 4;
  const int khalf = hs * 8, kb16 = hs * 16;
  const int r0 = wave * 16;   // this wave's 16 query rows

  // ---- S = Q @ K^T, K-dim = dd (padded to 64) ----
  v8f sacc[8];
  #pragma unroll
  for (int t = 0; t < 8; ++t)
    #pragma unroll
    for (int r = 0; r < 8; ++r) sacc[t][r] = 0.0f;

  #pragma unroll
  for (int kt = 0; kt < 64; kt += 32) {
    BFrag aq;   // A[n, dd] = Qs[n][dd]
    aq.q[0] = *(const uint4*)(&Qs[r0 + cm][kt + khalf]);
    aq.q[1] = *(const uint4*)(&Qs[r0 + cm][kt + khalf + 16]);
    #pragma unroll
    for (int t = 0; t < 8; ++t) {
      BFrag bk; // B[dd, m] = Ks[m][dd]
      bk.q[0] = *(const uint4*)(&Ks[t * 16 + cm][kt + khalf]);
      bk.q[1] = *(const uint4*)(&Ks[t * 16 + cm][kt + khalf + 16]);
      sacc[t] = __builtin_amdgcn_wmma_f32_16x16x32_bf16(
          false, aq.v, false, bk.v, (short)0, sacc[t], false, false);
    }
  }

  // ---- softmax over 128 cols (scale = 64^-0.5 = 0.125) ----
  float inv[8];
  #pragma unroll
  for (int r = 0; r < 8; ++r) {
    float mx = -1e30f;
    #pragma unroll
    for (int t = 0; t < 8; ++t) { sacc[t][r] *= 0.125f; mx = fmaxf(mx, sacc[t][r]); }
    #pragma unroll
    for (int d = 1; d < 16; d <<= 1) mx = fmaxf(mx, __shfl_xor(mx, d, 32));
    float sum = 0.0f;
    #pragma unroll
    for (int t = 0; t < 8; ++t) {
      float e = __expf(sacc[t][r] - mx);
      sacc[t][r] = e; sum += e;
    }
    #pragma unroll
    for (int d = 1; d < 16; d <<= 1) sum += __shfl_xor(sum, d, 32);
    inv[r] = 1.0f / sum;
  }

  __syncthreads();  // all waves done reading Qs/Ks before Ps overwrites them
  #pragma unroll
  for (int t = 0; t < 8; ++t)
    #pragma unroll
    for (int r = 0; r < 8; ++r)
      Ps[r0 + hs * 8 + r][t * 16 + cm] = (__bf16)(sacc[t][r] * inv[r]);
  __syncthreads();

  // ---- O = P @ V, K-dim = m (128), N = dd (48, valid 36) ----
  v8f oacc[3];
  #pragma unroll
  for (int nt = 0; nt < 3; ++nt)
    #pragma unroll
    for (int r = 0; r < 8; ++r) oacc[nt][r] = 0.0f;

  #pragma unroll
  for (int kt = 0; kt < 128; kt += 32) {
    BFrag ap;   // A[n, m] = Ps[n][m]
    ap.q[0] = *(const uint4*)(&Ps[r0 + cm][kt + khalf]);
    ap.q[1] = *(const uint4*)(&Ps[r0 + cm][kt + khalf + 16]);
    #pragma unroll
    for (int nt = 0; nt < 3; ++nt) {
      BFrag bv; // B[m, dd] = Vt[dd][m]
      bv.q[0] = *(const uint4*)(&Vt[nt * 16 + cm][kt + khalf]);
      bv.q[1] = *(const uint4*)(&Vt[nt * 16 + cm][kt + khalf + 16]);
      oacc[nt] = __builtin_amdgcn_wmma_f32_16x16x32_bf16(
          false, ap.v, false, bv.v, (short)0, oacc[nt], false, false);
    }
  }

  const long long obase = ((long long)b * 288 + h * 36) * 512 + s * 128;
  #pragma unroll
  for (int nt = 0; nt < 3; ++nt) {
    int dd = nt * 16 + cm;
    if (dd < 36) {
      #pragma unroll
      for (int r = 0; r < 8; ++r) {
        int n = r0 + hs * 8 + r;
        outp[obase + (long long)dd * 512 + n] = (__bf16)oacc[nt][r];
      }
    }
  }
}

// ---------------- launcher ----------------
extern "C" void kernel_launch(void* const* d_in, const int* in_sizes, int n_in,
                              void* d_out, int out_size, void* d_ws, size_t ws_size,
                              hipStream_t stream) {
  const float* x  = (const float*)d_in[0];
  const float* wr = (const float*)d_in[1];
  const float* br = (const float*)d_in[2];
  const float* wq = (const float*)d_in[3];
  const float* wo = (const float*)d_in[4];
  const float* bo = (const float*)d_in[5];
  const float* wc = (const float*)d_in[6];
  const float* bc = (const float*)d_in[7];
  float* out = (float*)d_out;

  char* ws = (char*)d_ws;
  __bf16* x_t    = (__bf16*)ws; ws += (size_t)16 * 2048 * 576 * 2;  // [b][l][c]
  __bf16* h_bf   = (__bf16*)ws; ws += (size_t)16 * 288 * 2048 * 2;  // [b][c][l]
  __bf16* qkv_bf = (__bf16*)ws; ws += (size_t)16 * 288 * 1536 * 2;  // [b][c][o]
  __bf16* at_bf  = (__bf16*)ws; ws += (size_t)16 * 288 * 512 * 2;   // [b][c][n]
  __bf16* o2_t   = (__bf16*)ws; ws += (size_t)16 * 2048 * 288 * 2;  // [b][l][c]
  __bf16* wr_bf  = (__bf16*)ws; ws += (size_t)288 * 576 * 2;
  __bf16* wq_bf  = (__bf16*)ws; ws += (size_t)1536 * 2048 * 2;
  __bf16* wo_bf  = (__bf16*)ws; ws += (size_t)2048 * 512 * 2;
  __bf16* wc_bf  = (__bf16*)ws; ws += (size_t)576 * 288 * 2;

  // x: [b][576][2048] f32 -> x_t [b][2048][576] bf16
  f2bf_transpose_kernel<<<dim3(64, 18, 16), 256, 0, stream>>>(x, x_t, 576, 2048);
  f2bf_kernel<<<256,  256, 0, stream>>>(wr, wr_bf, (long long)288 * 576);
  f2bf_kernel<<<1024, 256, 0, stream>>>(wq, wq_bf, (long long)1536 * 2048);
  f2bf_kernel<<<512,  256, 0, stream>>>(wo, wo_bf, (long long)2048 * 512);
  f2bf_kernel<<<256,  256, 0, stream>>>(wc, wc_bf, (long long)576 * 288);

  // 1) h = w_reduct @ x + b_reduct   (M=288,N=2048,K=576; B = x_t [N][K])
  gemm_bf16_wmma<1, false, __bf16><<<dim3(16, 3, 16), 256, 0, stream>>>(
      wr_bf, x_t, br, h_bf, 288, 2048, 576, 576, 576, 2048,
      0LL, 2048LL * 576, 288LL * 2048);
  // 2) qkv = h @ w_qkv^T             (M=288,N=1536,K=2048; B = w_qkv [N][K])
  gemm_bf16_wmma<0, false, __bf16><<<dim3(12, 3, 16), 256, 0, stream>>>(
      h_bf, wq_bf, nullptr, qkv_bf, 288, 1536, 2048, 2048, 2048, 1536,
      288LL * 2048, 0LL, 288LL * 1536);
  // 3) fused attention (512 blocks)
  attn_wmma_kernel<<<dim3(4, 8, 16), 256, 0, stream>>>(qkv_bf, at_bf);
  // 4) o2^T = (attn @ w_out^T + b_out)^T  stored [l][c] (M=288,N=2048,K=512)
  gemm_bf16_wmma<2, true, __bf16><<<dim3(16, 3, 16), 256, 0, stream>>>(
      at_bf, wo_bf, bo, o2_t, 288, 2048, 512, 512, 512, 288,
      288LL * 512, 0LL, 2048LL * 288);
  // 5) out = w_conv @ o2 + b_conv    (M=576,N=2048,K=288; B = o2_t [N][K]), fp32 out
  gemm_bf16_wmma<1, false, float><<<dim3(16, 5, 16), 256, 0, stream>>>(
      wc_bf, o2_t, bc, out, 576, 2048, 288, 288, 288, 2048,
      0LL, 2048LL * 288, 576LL * 2048);
}
